// RAALMultiHeadAttentionLayer_71012989272702
// MI455X (gfx1250) — matrix operand order repaired and checked by reading, not compiled, not run on hardware
//
#include <hip/hip_runtime.h>
#include <hip/hip_bf16.h>
#include <stdint.h>

// Problem constants from the reference
#define G_      1024
#define NNODES  64
#define E_      128
#define H_      8
#define D_      32
#define INDIM   256
#define NS_     48
#define HD      (H_*D_)      // 256
#define ROWS    (G_*NNODES)  // 65536
#define EPS_    1e-6f

typedef __attribute__((ext_vector_type(16))) __bf16 v16bf;
typedef __attribute__((ext_vector_type(8)))  __bf16 v8bf;
typedef __attribute__((ext_vector_type(8)))  float  v8f;
typedef __attribute__((ext_vector_type(4)))  float  v4f;

// ---------------------------------------------------------------------------
// Fragment loaders per CDNA5 ISA layouts (CDNA5 ISA §7.12.2)
// A 16x32 bf16: lane m / m+16: VGPR0-3 K=hi*8..hi*8+7, VGPR4-7 K=16+hi*8..
// B 32x16 bf16: lane n / n+16 holds column n; VGPR0-3 K=hi*8..hi*8+7, VGPR4-7 K=16+hi*8..
// ---------------------------------------------------------------------------
__device__ __forceinline__ v16bf load_frag_a_f32(const float* base, int ld, int lane) {
    int m  = lane & 15;
    int hi = lane >> 4;
    const float* p0 = base + m * ld + hi * 8;
    v16bf f;
#pragma unroll
    for (int t = 0; t < 8; ++t) {
        f[t]     = (__bf16)p0[t];
        f[8 + t] = (__bf16)p0[16 + t];
    }
    return f;
}

__device__ __forceinline__ v16bf load_frag_a_bf16(const __bf16* base, int ld, int lane) {
    int m  = lane & 15;
    int hi = lane >> 4;
    const __bf16* p0 = base + m * ld + hi * 8;
    v8bf lo = *(const v8bf*)(p0);        // 16B
    v8bf hh = *(const v8bf*)(p0 + 16);   // 16B
    v16bf f;
#pragma unroll
    for (int t = 0; t < 8; ++t) { f[t] = lo[t]; f[8 + t] = hh[t]; }
    return f;
}

__device__ __forceinline__ v16bf load_frag_b_bf16(const __bf16* base, int ld, int lane) {
    int n  = lane & 15;
    int hi = lane >> 4;
    // column n of B == row n of the (row-major, K-contiguous) source
    const __bf16* p0 = base + n * ld + hi * 8;
    v8bf lo = *(const v8bf*)(p0);        // 16B
    v8bf hh = *(const v8bf*)(p0 + 16);   // 16B
    v16bf f;
#pragma unroll
    for (int t = 0; t < 8; ++t) { f[t] = lo[t]; f[8 + t] = hh[t]; }
    return f;
}

// ---------------------------------------------------------------------------
// Kernel 0: transpose W (f32 256x256, K-major) -> WT (bf16, N-major rows)
// ---------------------------------------------------------------------------
__global__ __launch_bounds__(256) void prep_wt_kernel(
    const float* __restrict__ Wq, const float* __restrict__ Wk,
    const float* __restrict__ Wv, __bf16* __restrict__ WT)
{
    int t = blockIdx.x * 256 + threadIdx.x;          // 3*65536 total
    int m = t >> 16;
    int r = t & 65535;
    int n = r >> 8;
    int k = r & 255;
    const float* W = (m == 0) ? Wq : (m == 1) ? Wk : Wv;
    WT[(size_t)m * 65536 + n * 256 + k] = (__bf16)W[k * 256 + n];
}

// ---------------------------------------------------------------------------
// Kernel 1: QKV projection GEMM  (65536x256) x (256x256) + bias.
//   Data-flow: each wave owns 16 rows of h, converts its full 16x256 A stripe
//   to bf16 fragments ONCE (64 VGPRs), then streams the tiny cache-hot W^T
//   as B fragments: 1 B-load per WMMA, h read from HBM exactly once.
//   Each 16x16 f32 accumulator tile is repacked through per-wave LDS scratch
//   (col-major, stride 20 -> conflict-free, same-wave in-order so no barrier)
//   into coalesced b128 global stores.
// ---------------------------------------------------------------------------
__global__ __launch_bounds__(256) void qkv_gemm_kernel(
    const float* __restrict__ h, const __bf16* __restrict__ WT,
    const float* __restrict__ bq, const float* __restrict__ bk,
    const float* __restrict__ bv,
    __bf16* __restrict__ Qb, __bf16* __restrict__ Kb, float* __restrict__ Vf)
{
    __shared__ float repack[8][16 * 20];          // 10 KB, per-wave slices
    const int lane = threadIdx.x & 31;
    const int wave = threadIdx.x >> 5;            // 0..7
    const int row0 = (blockIdx.x * 8 + wave) * 16;

    // Load + convert the whole 16x256 A stripe once: 8 fragments.
    v16bf afrag[8];
#pragma unroll
    for (int kc = 0; kc < 8; ++kc)
        afrag[kc] = load_frag_a_f32(h + (size_t)row0 * INDIM + kc * 32, INDIM, lane);

    float* scratch = &repack[wave][0];
    const int n  = lane & 15;                     // D-tile column / out row
    const int hb = lane >> 4;

    for (int mat = 0; mat < 3; ++mat) {
        const __bf16* BT   = WT + (size_t)mat * 65536;
        const float*  bias = (mat == 0) ? bq : (mat == 1) ? bk : bv;
        for (int ct = 0; ct < 16; ++ct) {
            const int col0 = ct * 16;
            v8f acc = {};
#pragma unroll
            for (int kc = 0; kc < 8; ++kc) {
                v16bf b = load_frag_b_bf16(BT + (size_t)col0 * 256 + kc * 32, 256, lane);
                acc = __builtin_amdgcn_wmma_f32_16x16x32_bf16(
                    false, afrag[kc], false, b, (short)0, acc, false, false);
            }
            // bias + stash tile col-major into LDS (lane owns col n, rows hb*8..+7)
            {
                const float bn = bias[col0 + n];
                v4f lo, hi;
#pragma unroll
                for (int r = 0; r < 4; ++r) { lo[r] = acc[r] + bn; hi[r] = acc[r + 4] + bn; }
                *(v4f*)&scratch[n * 20 + hb * 8]     = lo;
                *(v4f*)&scratch[n * 20 + hb * 8 + 4] = hi;
            }
            // read back row-major (lane -> row n, cols hb*8..+7), coalesced store
            {
                const int cb = hb * 8;
                float vals[8];
#pragma unroll
                for (int c = 0; c < 8; ++c) vals[c] = scratch[(cb + c) * 20 + n];
                const size_t gidx = (size_t)(row0 + n) * HD + col0 + cb;
                if (mat < 2) {
                    __bf16* O = (mat == 0) ? Qb : Kb;
                    v8bf o;
#pragma unroll
                    for (int c = 0; c < 8; ++c) o[c] = (__bf16)vals[c];
                    *(v8bf*)(O + gidx) = o;                     // 16B store
                } else {
                    v4f o0, o1;
#pragma unroll
                    for (int c = 0; c < 4; ++c) { o0[c] = vals[c]; o1[c] = vals[c + 4]; }
                    *(v4f*)(Vf + gidx)     = o0;                // 16B store
                    *(v4f*)(Vf + gidx + 4) = o1;                // 16B store
                }
            }
        }
    }
}

// ---------------------------------------------------------------------------
// Kernel 2: per-(g,h) attention.
//   S = clip(Q_gh @ K_gh^T, +-5) via 16 WMMAs (K=D=32 -> one WMMA per tile),
//   V tile staged with CDNA5 async global->LDS copies (ASYNCcnt) overlapped
//   with the WMMAs + score gather, LDS-atomic segment sum, normalize.
//   128 threads = 4 waves; wave w owns S rows [16w, 16w+16).
// ---------------------------------------------------------------------------
__global__ __launch_bounds__(128) void attn_kernel(
    const __bf16* __restrict__ Qb, const __bf16* __restrict__ Kb,
    const float* __restrict__ Vf,
    const int* __restrict__ src, const int* __restrict__ dst,
    const int* __restrict__ ns, float* __restrict__ out)
{
    __shared__ float Ss[64][65];     // padded to dodge bank conflicts
    __shared__ float Vs[64][32];
    __shared__ float wVs[64][32];
    __shared__ float zs[64];
    __shared__ float score[E_];

    const int g    = blockIdx.y;
    const int hh   = blockIdx.x;
    const int tid  = threadIdx.x;
    const int lane = tid & 31;
    const int wave = tid >> 5;       // 0..3

    const __bf16* Qg = Qb + (size_t)(g * NNODES) * HD + hh * D_;
    const __bf16* Kg = Kb + (size_t)(g * NNODES) * HD + hh * D_;
    const float*  Vg = Vf + (size_t)(g * NNODES) * HD + hh * D_;

    // kick off async V-tile staging: 8 KB = 512 x b128, 4 per thread.
    // global_load_async_to_lds_b128: VDST = LDS byte address, VADDR = 64-bit
    // global address; tracked by ASYNCcnt (CDNA5 ISA).
#pragma unroll
    for (int it = 0; it < 4; ++it) {
        const int idx = tid + it * 128;          // 0..511
        const int nn  = idx >> 3;                // V row
        const int c4  = (idx & 7) * 4;           // float offset of 16B chunk
        const float* gp = Vg + (size_t)nn * HD + c4;
        const uint32_t lp = (uint32_t)(uintptr_t)&Vs[nn][c4];
        asm volatile("global_load_async_to_lds_b128 %0, %1, off"
                     :: "v"(lp), "v"(gp) : "memory");
    }

    // zero accumulators while the async copies fly
    for (int idx = tid; idx < NNODES * D_; idx += 128) wVs[idx >> 5][idx & 31] = 0.0f;
    if (tid < NNODES) zs[tid] = 0.0f;

    // S = Q K^T for 16 rows per wave; one WMMA per 16x16 tile (K=32)
    v16bf a = load_frag_a_bf16(Qg + (size_t)(16 * wave) * HD, HD, lane);
#pragma unroll
    for (int j = 0; j < 4; ++j) {
        v16bf b = load_frag_b_bf16(Kg + (size_t)(16 * j) * HD, HD, lane);
        v8f acc = {};
        acc = __builtin_amdgcn_wmma_f32_16x16x32_bf16(false, a, false, b, (short)0, acc, false, false);
        const int nc = lane & 15;
        const int hb = lane >> 4;
#pragma unroll
        for (int r = 0; r < 8; ++r) {
            float v = acc[r];
            v = fminf(fmaxf(v, -5.0f), 5.0f);
            Ss[16 * wave + hb * 8 + r][16 * j + nc] = v;
        }
    }
    __syncthreads();

    // per-edge score: num / (sum over sampled neighbors + eps)
    {
        const int e = tid;           // E_ == 128 == blockDim.x
        const int r = src[e];
        const float num = Ss[r][dst[e]];
        float den = 0.0f;
        const int* nse = ns + e * NS_;
#pragma unroll 8
        for (int j = 0; j < NS_; ++j) den += Ss[r][nse[j]];
        score[e] = num / (den + EPS_);
    }
    // drain our async V copies, then make them block-visible
    asm volatile("s_wait_asynccnt 0" ::: "memory");
    __syncthreads();

    // segment-sum: 4 edges in flight, 32 lanes (one per d) each
    for (int eb = 0; eb < E_; eb += 4) {
        const int e  = eb + (tid >> 5);
        const int d  = tid & 31;
        const float sc = score[e];
        const int   dn = dst[e];
        atomicAdd(&wVs[dn][d], Vs[src[e]][d] * sc);
        if (d == 0) atomicAdd(&zs[dn], sc);
    }
    __syncthreads();

    // normalize + store
    for (int idx = tid; idx < NNODES * D_; idx += 128) {
        int nn = idx >> 5, dd = idx & 31;
        out[(size_t)(g * NNODES + nn) * HD + hh * D_ + dd] = wVs[nn][dd] / (zs[nn] + EPS_);
    }
}

// ---------------------------------------------------------------------------
extern "C" void kernel_launch(void* const* d_in, const int* in_sizes, int n_in,
                              void* d_out, int out_size, void* d_ws, size_t ws_size,
                              hipStream_t stream) {
    (void)in_sizes; (void)n_in; (void)out_size; (void)ws_size;
    const float* h   = (const float*)d_in[0];
    const float* Wq  = (const float*)d_in[1];
    const float* bq  = (const float*)d_in[2];
    const float* Wk  = (const float*)d_in[3];
    const float* bk  = (const float*)d_in[4];
    const float* Wv  = (const float*)d_in[5];
    const float* bv  = (const float*)d_in[6];
    const int*   src = (const int*)d_in[7];
    const int*   dst = (const int*)d_in[8];
    const int*   ns  = (const int*)d_in[9];
    float* out = (float*)d_out;

    // workspace layout (total ~134.7 MB)
    char* ws = (char*)d_ws;
    __bf16* WT = (__bf16*)ws;                                        // 384 KB
    __bf16* Qb = (__bf16*)(ws + (512u << 10));                       // 32 MB
    __bf16* Kb = (__bf16*)(ws + (512u << 10) + 33554432u);           // 32 MB
    float*  Vf = (float*)(ws + (512u << 10) + 2u * 33554432u);       // 64 MB

    prep_wt_kernel<<<768, 256, 0, stream>>>(Wq, Wk, Wv, WT);
    qkv_gemm_kernel<<<ROWS / 128, 256, 0, stream>>>(h, WT, bq, bk, bv, Qb, Kb, Vf);
    attn_kernel<<<dim3(H_, G_), 128, 0, stream>>>(Qb, Kb, Vf, src, dst, ns, out);
}
